// DifferentiableMemory_61452392071326
// MI455X (gfx1250) — compile-verified, matrix operand order checked by reading, not run on hardware
//
#include <hip/hip_runtime.h>
#include <hip/hip_bf16.h>
#include <math.h>

// ---------------------------------------------------------------------------
// Fused "DifferentiableMemory" forward for MI455X (gfx1250, wave32, WMMA).
// v2: pre-swizzled bf16 B-fragments in workspace, 64 rows/block (8 waves),
//     GEMM1 weight K-panels double-buffered into LDS with
//     global_load_async_to_lds_b128 (ASYNCcnt) shared by all M-waves.
// ---------------------------------------------------------------------------

typedef __attribute__((ext_vector_type(16))) __bf16 v16bf;
typedef __attribute__((ext_vector_type(8)))  __bf16 v8bf;
typedef __attribute__((ext_vector_type(8)))  float  v8f;

#define B_TOT   32768
#define D_IN    768
#define H1      256
#define E_DIM   128
#define N_CENT  500
#define N_PAD   512
#define TOT     902
#define TOT_PAD 928     // 29 * 32
#define K_TOP   5

#define ROWS     64     // rows of B per block
#define NTHREADS 256    // 8 waves: 4 M-tiles x 2 N-groups
#define NK1      24     // 768 / 32 K-steps for GEMM1

// ---- LDS layout (bytes) ----------------------------------------------------
#define PANEL_BYTES 16384                    // one 32x256 bf16 K-panel of enc_w1
#define OFF_PANEL   0                        // 2 * 16384 = 32768 (double buffer)
#define OFF_COMB    32768                    // 64 x 928 bf16 = 118784
#define OFF_ENCB    151552                   // 64 x 128 bf16 = 16384
#define OFF_HID     167936                   // 64 x 256 bf16 = 32768  } union
#define OFF_SIMS    167936                   // 64 x 512 bf16 = 65536  } (disjoint lifetime)
#define OFF_IMPH    233472                   // 64 x 64 f32   = 16384
#define OFF_NORM    249856                   // 64 f32
#define OFF_EMO     250112                   // 64 f32
#define SMEM_BYTES  250368

// ---- workspace layout (bytes) ---------------------------------------------
#define WS_CNORM 0          // 512 f32
#define WS_W1F   2048       // 768*256  bf16 = 393216
#define WS_W2F   395264     // 256*128  bf16 = 65536
#define WS_IMPF  460800     // 928*64   bf16 = 118784
#define WS_CENTF 579584     // 128*512  bf16 = 131072  (B = centroids^T, N padded)

static __device__ inline float gelu_exact(float x) {
    return 0.5f * x * (1.0f + erff(x * 0.70710678118654752f));
}

static __device__ inline v8f wmma_bf16(v16bf a, v16bf b, v8f c) {
    return __builtin_amdgcn_wmma_f32_16x16x32_bf16(false, a, false, b, (short)0, c,
                                                   false, false);
}

// A-fragment from LDS (bf16 row-major, 16B-aligned rows).
// lanes 0-15: M=lane,    K {k0+0..7,  k0+16..23}
// lanes16-31: M=lane-16, K {k0+8..15, k0+24..31}
static __device__ inline v16bf load_a_lds(const __bf16* base, int ld, int m,
                                          int k0, int koff) {
    const v8bf* p0 = (const v8bf*)(base + m * ld + k0 + koff);
    const v8bf* p1 = (const v8bf*)(base + m * ld + k0 + 16 + koff);
    return __builtin_shufflevector(*p0, *p1,
                                   0, 1, 2, 3, 4, 5, 6, 7,
                                   8, 9, 10, 11, 12, 13, 14, 15);
}

// B-fragment from pre-swizzled bf16 array: [ks][nt][lane][16] (32B/lane).
static __device__ inline v16bf load_frag_g(const __bf16* base, int ks, int NT,
                                           int nt, int lane) {
    return *(const v16bf*)(base + (((size_t)(ks * NT + nt) * 32 + lane) << 4));
}

// B-fragment from LDS panel: [nt][lane][16].
static __device__ inline v16bf load_frag_l(const __bf16* pbase, int nt, int lane) {
    return *(const v16bf*)(pbase + (((nt << 5) + lane) << 4));
}

// gfx1250 async copy: 16 bytes/lane, global (SGPR base + VGPR offset) -> LDS.
static __device__ inline void async_copy_b128(unsigned lds_off, unsigned long long gbase,
                                              unsigned goff) {
    asm volatile("global_load_async_to_lds_b128 %0, %1, %2"
                 :: "v"(lds_off), "v"(goff), "s"(gbase) : "memory");
}

// Issue one 16KB enc_w1 K-panel (panel p -> LDS buffer p&1). 256 threads x
// 4 b128 issues cover the whole panel; 4 ASYNCcnt increments per wave.
static __device__ inline void issue_panel(char* smem, unsigned long long gbase,
                                          int p, int tid) {
    unsigned tbyte = (unsigned)tid * 16u;
    unsigned lds0 = (unsigned)(uintptr_t)(smem + OFF_PANEL) +
                    (unsigned)((p & 1) * PANEL_BYTES) + tbyte;
    unsigned g0 = (unsigned)p * (unsigned)PANEL_BYTES + tbyte;
#pragma unroll
    for (int ii = 0; ii < 4; ++ii)
        async_copy_b128(lds0 + ii * 4096u, gbase, g0 + ii * 4096u);
}

// ---------------------------------------------------------------------------
__global__ void centroid_norm_kernel(const float* __restrict__ centroids,
                                     float* __restrict__ c_norm) {
    int n = blockIdx.x * blockDim.x + threadIdx.x;
    if (n < N_PAD) {
        float s = 0.0f;
        if (n < N_CENT) {
            const float* row = centroids + n * E_DIM;
#pragma unroll 8
            for (int c = 0; c < E_DIM; ++c) { float v = row[c]; s += v * v; }
            s = sqrtf(s);
        }
        c_norm[n] = s;
    }
}

// Pre-swizzle a B operand into wave32 bf16 fragment order: dst[ks][nt][lane][16].
// transposed==0: B[k][n] = src[k*ld + n];  transposed==1: B[k][n] = src[n*ld + k].
__global__ void swizzle_b_kernel(const float* __restrict__ src, int ld, int kmax,
                                 int nmax, int NT, int total, int transposed,
                                 __bf16* __restrict__ dst) {
    int idx = blockIdx.x * blockDim.x + threadIdx.x;
    if (idx >= total) return;
    int i    = idx & 15;
    int lane = (idx >> 4) & 31;
    int nt   = (idx >> 9) % NT;
    int ks   = (idx >> 9) / NT;
    int koff = (lane < 16) ? 0 : 8;
    int n = nt * 16 + (lane & 15);
    int k = ks * 32 + ((i < 8) ? (koff + i) : (16 + koff + (i - 8)));
    float v = 0.0f;
    if (k < kmax && n < nmax)
        v = transposed ? src[n * ld + k] : src[k * ld + n];
    dst[idx] = (__bf16)v;
}

// ---------------------------------------------------------------------------
__global__ __launch_bounds__(NTHREADS)
void fused_memory_kernel(const float* __restrict__ cue,
                         const float* __restrict__ internal_state,
                         const float* __restrict__ reward,
                         const float* __restrict__ timestamp,
                         const float* __restrict__ emotional,
                         const float* __restrict__ enc_b1,
                         const float* __restrict__ enc_b2,
                         const float* __restrict__ imp_b1,
                         const float* __restrict__ imp_w2,
                         const float* __restrict__ imp_b2,
                         const __bf16* __restrict__ w1f,
                         const __bf16* __restrict__ w2f,
                         const __bf16* __restrict__ impf,
                         const __bf16* __restrict__ centf,
                         const float* __restrict__ c_norm,
                         float* __restrict__ out) {
    extern __shared__ char smem[];
    __bf16* sComb = (__bf16*)(smem + OFF_COMB);   // [64][928]
    __bf16* sHid  = (__bf16*)(smem + OFF_HID);    // [64][256]
    __bf16* sEncB = (__bf16*)(smem + OFF_ENCB);   // [64][128]
    __bf16* sSims = (__bf16*)(smem + OFF_SIMS);   // [64][512]
    float*  sImpH = (float*)(smem + OFF_IMPH);    // [64][64]
    float*  sNorm = (float*)(smem + OFF_NORM);    // [64]
    float*  sEmo  = (float*)(smem + OFF_EMO);     // [64]

    const int tid  = threadIdx.x;
    const int wave = tid >> 5;
    const int lane = tid & 31;
    const int mw   = wave & 3;      // M-tile 0..3 (16 rows each)
    const int nw   = wave >> 2;     // N-group 0..1
    const int row0 = blockIdx.x * ROWS;
    const int nlo  = lane & 15;
    const int koff = (lane < 16) ? 0 : 8;
    const int mhi  = (lane < 16) ? 0 : 8;
    const int arow = mw * 16 + nlo; // A-matrix row this lane reads

    // ---------------- Stage A: stage combined = [cue | state | r | t | emo | 0pad]
    {
        const float4* cue4 = (const float4*)(cue + (size_t)row0 * D_IN);
        for (int i = tid; i < ROWS * D_IN / 4; i += NTHREADS) {
            float4 v = cue4[i];
            int r = (i * 4) / D_IN, c = (i * 4) % D_IN;
            __bf16* dst = sComb + r * TOT_PAD + c;
            dst[0] = (__bf16)v.x; dst[1] = (__bf16)v.y;
            dst[2] = (__bf16)v.z; dst[3] = (__bf16)v.w;
        }
        const float4* is4 = (const float4*)(internal_state + (size_t)row0 * E_DIM);
        for (int i = tid; i < ROWS * E_DIM / 4; i += NTHREADS) {
            float4 v = is4[i];
            int r = (i * 4) / E_DIM, c = (i * 4) % E_DIM;
            __bf16* dst = sComb + r * TOT_PAD + D_IN + c;
            dst[0] = (__bf16)v.x; dst[1] = (__bf16)v.y;
            dst[2] = (__bf16)v.z; dst[3] = (__bf16)v.w;
        }
        if (tid < ROWS) {
            int r = tid, g = row0 + r;
            sComb[r * TOT_PAD + 896] = (__bf16)reward[g];
            sComb[r * TOT_PAD + 897] = (__bf16)timestamp[g];
            float es = 0.0f;
#pragma unroll
            for (int j = 0; j < 4; ++j) {
                float e = emotional[g * 4 + j];
                es += e;
                sComb[r * TOT_PAD + 898 + j] = (__bf16)e;
            }
            sEmo[r] = es * 0.25f;
            for (int c = TOT; c < TOT_PAD; ++c)
                sComb[r * TOT_PAD + c] = (__bf16)0.0f;
        }
        __builtin_prefetch(w2f + lane * 16, 0, 0);   // warm L2 for stage C
    }
    issue_panel(smem, (unsigned long long)(uintptr_t)w1f, 0, tid);   // panel 0 in flight
    __syncthreads();

    // ---------------- Stage B: h1 = gelu(cue @ enc_w1 + b1)   [64 x 256]
    {
        v8f acc[8];
#pragma unroll
        for (int t = 0; t < 8; ++t) {
            float bias = enc_b1[(nw * 8 + t) * 16 + nlo];
#pragma unroll
            for (int j = 0; j < 8; ++j) acc[t][j] = bias;
        }
        for (int ks = 0; ks < NK1; ++ks) {
            if (ks + 1 < NK1) {
                issue_panel(smem, (unsigned long long)(uintptr_t)w1f, ks + 1, tid);
                asm volatile("s_wait_asynccnt 4" ::: "memory");   // panel ks landed
            } else {
                asm volatile("s_wait_asynccnt 0" ::: "memory");
            }
            __syncthreads();                                      // panel ks visible
            const __bf16* pbase = (const __bf16*)(smem + OFF_PANEL +
                                                  (ks & 1) * PANEL_BYTES);
            v16bf a = load_a_lds(sComb, TOT_PAD, arow, ks * 32, koff);
#pragma unroll
            for (int t = 0; t < 8; ++t) {
                v16bf b = load_frag_l(pbase, nw * 8 + t, lane);
                acc[t] = wmma_bf16(a, b, acc[t]);
            }
            __syncthreads();                  // buffer (ks&1) free to overwrite
        }
#pragma unroll
        for (int t = 0; t < 8; ++t) {
            int n = (nw * 8 + t) * 16 + nlo;
#pragma unroll
            for (int j = 0; j < 8; ++j)
                sHid[(mw * 16 + j + mhi) * H1 + n] = (__bf16)gelu_exact(acc[t][j]);
        }
    }
    __syncthreads();

    // ---------------- Stage C: enc = h1 @ enc_w2 + b2         [64 x 128]
    {
        v8f acc[4];
#pragma unroll
        for (int t = 0; t < 4; ++t) {
            float bias = enc_b2[(nw * 4 + t) * 16 + nlo];
#pragma unroll
            for (int j = 0; j < 8; ++j) acc[t][j] = bias;
        }
        for (int ks = 0; ks < H1 / 32; ++ks) {
            v16bf a = load_a_lds(sHid, H1, arow, ks * 32, koff);
#pragma unroll
            for (int t = 0; t < 4; ++t) {
                v16bf b = load_frag_g(w2f, ks, 8, nw * 4 + t, lane);
                acc[t] = wmma_bf16(a, b, acc[t]);
            }
        }
#pragma unroll
        for (int t = 0; t < 4; ++t) {
            int n = (nw * 4 + t) * 16 + nlo;
#pragma unroll
            for (int j = 0; j < 8; ++j)
                sEncB[(mw * 16 + j + mhi) * E_DIM + n] = (__bf16)acc[t][j];
        }
    }
    __syncthreads();

    if (tid < ROWS) {
        float s = 0.0f;
#pragma unroll 8
        for (int c = 0; c < E_DIM; ++c) {
            float v = (float)sEncB[tid * E_DIM + c];
            s += v * v;
        }
        sNorm[tid] = sqrtf(s);
    }
    __syncthreads();   // also: sHid reads done -> sSims may overwrite that region

    // ---------------- Stage D: sims = (enc @ cent^T) / max(|e||c|, eps)  [64 x 512]
    for (int t = 0; t < 16; ++t) {
        const int nt = nw * 16 + t;
        const int n  = nt * 16 + nlo;
        v8f acc;
#pragma unroll
        for (int j = 0; j < 8; ++j) acc[j] = 0.0f;
#pragma unroll
        for (int ks = 0; ks < E_DIM / 32; ++ks) {
            v16bf a = load_a_lds(sEncB, E_DIM, arow, ks * 32, koff);
            v16bf b = load_frag_g(centf, ks, 32, nt, lane);
            acc = wmma_bf16(a, b, acc);
        }
        float cn = c_norm[n];
#pragma unroll
        for (int j = 0; j < 8; ++j) {
            int m = mw * 16 + j + mhi;
            float denom = fmaxf(sNorm[m] * cn, 1e-8f);
            float v = (n < N_CENT) ? acc[j] / denom : -1.0e30f;
            sSims[m * N_PAD + n] = (__bf16)v;
        }
    }

    // ---------------- Stage E: imp hidden = gelu(combined @ imp_w1 + b1) [64 x 64]
    {
        v8f acc[2];
#pragma unroll
        for (int t = 0; t < 2; ++t) {
            float bias = imp_b1[(nw * 2 + t) * 16 + nlo];
#pragma unroll
            for (int j = 0; j < 8; ++j) acc[t][j] = bias;
        }
        for (int ks = 0; ks < TOT_PAD / 32; ++ks) {
            v16bf a = load_a_lds(sComb, TOT_PAD, arow, ks * 32, koff);
#pragma unroll
            for (int t = 0; t < 2; ++t) {
                v16bf b = load_frag_g(impf, ks, 4, nw * 2 + t, lane);
                acc[t] = wmma_bf16(a, b, acc[t]);
            }
        }
#pragma unroll
        for (int t = 0; t < 2; ++t) {
            int n = (nw * 2 + t) * 16 + nlo;
#pragma unroll
            for (int j = 0; j < 8; ++j)
                sImpH[(mw * 16 + j + mhi) * 64 + n] = gelu_exact(acc[t][j]);
        }
    }
    __syncthreads();

    // ---------------- Stage F: top-5 + importance head, write [64 x 6]
    if (tid < ROWS) {
        int r = tid, g = row0 + r;
        float s = imp_b2[0];
#pragma unroll 8
        for (int c = 0; c < 64; ++c) s += sImpH[r * 64 + c] * imp_w2[c];
        float imp = (1.0f / (1.0f + expf(-s))) * sEmo[r];

        float tk[K_TOP];
#pragma unroll
        for (int j = 0; j < K_TOP; ++j) tk[j] = -3.0e38f;
        const __bf16* sr = sSims + r * N_PAD;
        for (int c = 0; c < N_PAD; ++c) {
            float v = (float)sr[c];
            if (v > tk[K_TOP - 1]) {
                int p = K_TOP - 1;
                while (p > 0 && tk[p - 1] < v) { tk[p] = tk[p - 1]; --p; }
                tk[p] = v;
            }
        }
        float* o = out + (size_t)g * (K_TOP + 1);
#pragma unroll
        for (int j = 0; j < K_TOP; ++j) o[j] = tk[j];
        o[K_TOP] = imp;
    }
}

// ---------------------------------------------------------------------------
extern "C" void kernel_launch(void* const* d_in, const int* in_sizes, int n_in,
                              void* d_out, int out_size, void* d_ws, size_t ws_size,
                              hipStream_t stream) {
    const float* cue            = (const float*)d_in[0];
    const float* internal_state = (const float*)d_in[1];
    const float* reward         = (const float*)d_in[2];
    const float* timestamp      = (const float*)d_in[3];
    const float* emotional      = (const float*)d_in[4];
    const float* centroids      = (const float*)d_in[5];
    const float* enc_w1         = (const float*)d_in[6];
    const float* enc_b1         = (const float*)d_in[7];
    const float* enc_w2         = (const float*)d_in[8];
    const float* enc_b2         = (const float*)d_in[9];
    const float* imp_w1         = (const float*)d_in[10];
    const float* imp_b1         = (const float*)d_in[11];
    const float* imp_w2         = (const float*)d_in[12];
    const float* imp_b2         = (const float*)d_in[13];
    float*       out            = (float*)d_out;

    char* ws = (char*)d_ws;
    float*  c_norm = (float*)(ws + WS_CNORM);
    __bf16* w1f    = (__bf16*)(ws + WS_W1F);
    __bf16* w2f    = (__bf16*)(ws + WS_W2F);
    __bf16* impf   = (__bf16*)(ws + WS_IMPF);
    __bf16* centf  = (__bf16*)(ws + WS_CENTF);

    centroid_norm_kernel<<<(N_PAD + 255) / 256, 256, 0, stream>>>(centroids, c_norm);

    // Pre-swizzle B operands into wave32 bf16 fragment order.
    int t1 = D_IN * H1;          // 196608
    int t2 = H1 * E_DIM;         // 32768
    int t3 = TOT_PAD * 64;       // 59392
    int t4 = E_DIM * N_PAD;      // 65536
    swizzle_b_kernel<<<(t1 + 255) / 256, 256, 0, stream>>>(enc_w1, H1, D_IN, H1, 16, t1, 0, w1f);
    swizzle_b_kernel<<<(t2 + 255) / 256, 256, 0, stream>>>(enc_w2, E_DIM, H1, E_DIM, 8, t2, 0, w2f);
    swizzle_b_kernel<<<(t3 + 255) / 256, 256, 0, stream>>>(imp_w1, 64, TOT, 64, 4, t3, 0, impf);
    swizzle_b_kernel<<<(t4 + 255) / 256, 256, 0, stream>>>(centroids, E_DIM, E_DIM, N_CENT, 32, t4, 1, centf);

    fused_memory_kernel<<<B_TOT / ROWS, NTHREADS, SMEM_BYTES, stream>>>(
        cue, internal_state, reward, timestamp, emotional,
        enc_b1, enc_b2, imp_b1, imp_w2, imp_b2,
        w1f, w2f, impf, centf, c_norm, out);
}
